// MultiHeadAttention_57629871178274
// MI455X (gfx1250) — matrix operand order repaired and checked by reading, not compiled, not run on hardware
//
#include <hip/hip_runtime.h>
#include <hip/hip_bf16.h>
#include <stdint.h>

// ---------------------------------------------------------------------------
// Types for CDNA5 WMMA bf16 path
// ---------------------------------------------------------------------------
typedef __attribute__((ext_vector_type(16))) __bf16 bf16x16;
typedef __attribute__((ext_vector_type(8)))  float  f32x8;

union Frag {
    uint4    u[2];
    bf16x16  v;
};

static __device__ __forceinline__ uint16_t f2bf(float f) {
    union { float f; uint32_t u; } x; x.f = f;
    uint32_t r = x.u + 0x7FFFu + ((x.u >> 16) & 1u);   // round-to-nearest-even
    return (uint16_t)(r >> 16);
}
static __device__ __forceinline__ float bf2f(uint16_t b) {
    union { uint32_t u; float f; } x; x.u = ((uint32_t)b) << 16;
    return x.f;
}

template<int XORM>
static __device__ __forceinline__ float swz_xor(float x) {
    int i = __builtin_bit_cast(int, x);
    i = __builtin_amdgcn_ds_swizzle(i, (XORM << 10) | 0x1f);  // group-of-32, xor mask
    return __builtin_bit_cast(float, i);
}

#define WMMA_BF16(A, B, C) \
    __builtin_amdgcn_wmma_f32_16x16x32_bf16(false, (A), false, (B), (short)0, (C), false, false)

// ---------------------------------------------------------------------------
// Async global -> LDS (CDNA5 ASYNCcnt path), with safe fallback.
// Probe result (round 2): param0 is generic GCC-vector 'int4 *'.
// ---------------------------------------------------------------------------
typedef int gvec4i __attribute__((vector_size(16)));

#if __has_builtin(__builtin_amdgcn_global_load_async_to_lds_b128) && \
    __has_builtin(__builtin_amdgcn_s_wait_asynccnt)
#define ASYNC_LDS 1
static __device__ __forceinline__ void g2l_b128(const void* g, void* l) {
    // LDS byte offset == low 32 bits of the flat LDS address (aperture layout)
    __builtin_amdgcn_global_load_async_to_lds_b128(
        (gvec4i*)g,
        (__attribute__((address_space(3))) gvec4i*)(uint32_t)(uintptr_t)l,
        0, 0);
}
static __device__ __forceinline__ void wait_async0() {
    __builtin_amdgcn_s_wait_asynccnt(0);
}
#else
#define ASYNC_LDS 0
static __device__ __forceinline__ void g2l_b128(const void* g, void* l) {
    *(uint4*)l = *(const uint4*)g;
}
static __device__ __forceinline__ void wait_async0() {}
#endif

// Compile-time memory scheduling barrier (same-wave DS ops are in-order in HW)
static __device__ __forceinline__ void sched_mem_fence() {
    asm volatile("" ::: "memory");
}

// Problem constants
#define BATCH   2
#define SEQ     2048
#define DMODEL  2048
#define NHEADS  16
#define DHEAD   128
#define MROWS   (BATCH * SEQ)   // 4096

// ---------------------------------------------------------------------------
// fp32 -> bf16 conversion
// ---------------------------------------------------------------------------
__global__ void cvt_f32_bf16(const float* __restrict__ in, uint16_t* __restrict__ out, int n) {
    int i = blockIdx.x * 256 + threadIdx.x;
    if (i < n) out[i] = f2bf(in[i]);
}

// ---------------------------------------------------------------------------
// NT GEMM: C[M,N] = A[M,K] * B[N,K]^T   (A,B bf16 row-major, K contiguous)
// Block tile 128x128, 8 waves (2x4), wave tile 64x32.
// K-slab 64, double-buffered LDS, async global->LDS prefetch.
// EPI==0: fp32 store to Cf[M,N].  EPI==1: bf16 scatter into QKV[B,3,H,S,Dh].
// ---------------------------------------------------------------------------
template<int EPI>
__global__ __launch_bounds__(256) void gemm_bf16_nt(
    const uint16_t* __restrict__ A, const uint16_t* __restrict__ Bw,
    float* __restrict__ Cf, uint16_t* __restrict__ Cq,
    int M, int N, int K)
{
    __shared__ uint16_t As[2][128][72];   // 64 K + 8 pad (row stride 144B, 16B aligned)
    __shared__ uint16_t Bs[2][128][72];

    const int tid  = threadIdx.x;
    const int wave = tid >> 5, lane = tid & 31;
    const int half = lane >> 4, ln = lane & 15;
    const int wm   = wave >> 2, wn = wave & 3;
    const int bm   = blockIdx.y, bn = blockIdx.x;

    f32x8 acc[4][2];
    #pragma unroll
    for (int i = 0; i < 4; ++i)
        #pragma unroll
        for (int j = 0; j < 2; ++j)
            #pragma unroll
            for (int e = 0; e < 8; ++e) acc[i][j][e] = 0.0f;

    const int lr = tid >> 1;            // 0..127 : row loaded by this thread
    const int lc = (tid & 1) * 32;      // 0 or 32: starting K element
    const uint16_t* aRow = A  + (size_t)(bm * 128 + lr) * K + lc;
    const uint16_t* bRow = Bw + (size_t)(bn * 128 + lr) * K + lc;

    auto stage = [&](int k0, int buf) {
        #pragma unroll
        for (int c = 0; c < 4; ++c) {
            g2l_b128(aRow + k0 + c * 8, &As[buf][lr][lc + c * 8]);
            g2l_b128(bRow + k0 + c * 8, &Bs[buf][lr][lc + c * 8]);
        }
    };

    stage(0, 0);
    for (int k0 = 0; k0 < K; k0 += 64) {
        const int p = (k0 >> 6) & 1;
        wait_async0();
        __syncthreads();                 // buffer p ready for everyone
        if (k0 + 64 < K) stage(k0 + 64, p ^ 1);   // prefetch overlaps compute

        #pragma unroll
        for (int kk = 0; kk < 64; kk += 32) {
            Frag af[4], bf[2];
            #pragma unroll
            for (int i = 0; i < 4; ++i) {
                const uint16_t* pp = &As[p][wm * 64 + i * 16 + ln][kk + half * 8];
                af[i].u[0] = *(const uint4*)pp;
                af[i].u[1] = *(const uint4*)(pp + 16);
            }
            #pragma unroll
            for (int j = 0; j < 2; ++j) {
                const uint16_t* pp = &Bs[p][wn * 32 + j * 16 + ln][kk + half * 8];
                bf[j].u[0] = *(const uint4*)pp;
                bf[j].u[1] = *(const uint4*)(pp + 16);
            }
            #pragma unroll
            for (int i = 0; i < 4; ++i)
                #pragma unroll
                for (int j = 0; j < 2; ++j)
                    acc[i][j] = WMMA_BF16(af[i].v, bf[j].v, acc[i][j]);
        }
        // no bottom barrier: next iteration's top barrier orders reads of buffer p
        // against the stage() that overwrites it right after that barrier
    }

    // Epilogue: C/D layout -> lane holds col ln, VGPR r holds row r + 8*half
    #pragma unroll
    for (int i = 0; i < 4; ++i)
        #pragma unroll
        for (int j = 0; j < 2; ++j)
            #pragma unroll
            for (int r = 0; r < 8; ++r) {
                int row = bm * 128 + wm * 64 + i * 16 + half * 8 + r;
                int col = bn * 128 + wn * 32 + j * 16 + ln;
                float v = acc[i][j][r];
                if (EPI == 0) {
                    Cf[(size_t)row * N + col] = v;
                } else {
                    int b = row >> 11, s = row & 2047;      // row = b*SEQ + s
                    int which = col >> 11;                   // 0=q 1=k 2=v
                    int h  = (col >> 7) & 15;
                    int dh = col & 127;
                    size_t dst = ((((size_t)b * 3 + which) * NHEADS + h) * SEQ + s) * DHEAD + dh;
                    Cq[dst] = f2bf(v);
                }
            }
}

// ---------------------------------------------------------------------------
// RoPE on q,k (in place, bf16)
// ---------------------------------------------------------------------------
__global__ void rope_kernel(uint16_t* __restrict__ qkv, const int* __restrict__ start_pos) {
    int idx = blockIdx.x * 256 + threadIdx.x;           // 2^23 total
    int i     = idx & 63;
    int s     = (idx >> 6) & 2047;
    int h     = (idx >> 17) & 15;
    int which = (idx >> 21) & 1;                        // q or k
    int b     = idx >> 22;
    size_t base = ((((size_t)b * 3 + which) * NHEADS + h) * SEQ + s) * DHEAD;
    float pos = (float)(s + start_pos[0]);
    float ang = pos * __expf(-0.14391156f * (float)i);  // 10000^(-i/64)
    float sn, cs;
    __sincosf(ang, &sn, &cs);
    float t1 = bf2f(qkv[base + i]);
    float t2 = bf2f(qkv[base + i + 64]);
    qkv[base + i]      = f2bf(t1 * cs - t2 * sn);
    qkv[base + i + 64] = f2bf(t2 * cs + t1 * sn);
}

// V transpose: v[B,H,S,Dh] -> vT[B,H,Dh,S]
__global__ void vtrans_kernel(const uint16_t* __restrict__ qkv, uint16_t* __restrict__ vT) {
    int idx = blockIdx.x * 256 + threadIdx.x;           // 2^23 total
    int dh = idx & 127;
    int s  = (idx >> 7) & 2047;
    int h  = (idx >> 18) & 15;
    int b  = idx >> 22;
    size_t src = ((((size_t)b * 3 + 2) * NHEADS + h) * SEQ + s) * DHEAD + dh;
    size_t dst = (((size_t)b * NHEADS + h) * DHEAD + dh) * SEQ + s;
    vT[dst] = qkv[src];
}

// ---------------------------------------------------------------------------
// Flash attention (causal). One block = (b, h, 128 query rows); 8 waves,
// each wave owns 16 query rows. K and V^T blocks staged cooperatively in
// double-buffered LDS via async global->LDS; P re-layout via per-wave LDS.
// ---------------------------------------------------------------------------
__global__ __launch_bounds__(256) void flash_attn_kernel(
    const uint16_t* __restrict__ qkv, const uint16_t* __restrict__ vT,
    uint16_t* __restrict__ om)
{
    __shared__ uint16_t Ks[2][128][136];   // keys x d      (+8 pad)
    __shared__ uint16_t Vs[2][128][136];   // dh   x keys   (+8 pad)
    __shared__ uint16_t Pl[8][16][136];    // per-wave 16x128 P tile

    const int tid  = threadIdx.x;
    const int wave = tid >> 5, lane = tid & 31;
    const int half = lane >> 4, ln = lane & 15;
    const int blk  = blockIdx.x;          // 512 blocks
    const int qblk = blk & 15;
    const int h    = (blk >> 4) & 15;
    const int b    = blk >> 8;

    const size_t qoff = (((size_t)b * 3 + 0) * NHEADS + h) * (size_t)SEQ * DHEAD;
    const size_t koff = (((size_t)b * 3 + 1) * NHEADS + h) * (size_t)SEQ * DHEAD;
    const size_t voff = (((size_t)b * NHEADS + h) * DHEAD) * (size_t)SEQ;
    const int q0 = qblk * 128 + wave * 16;

    // Cooperative staging: thread -> (row = tid>>1, 64-element half = (tid&1)*64)
    const int srow = tid >> 1;
    const int scol = (tid & 1) * 64;
    auto stageKV = [&](int kb, int buf) {
        const int kbase = kb * 128;
        const uint16_t* kg = qkv + koff + (size_t)(kbase + srow) * DHEAD + scol;
        const uint16_t* vg = vT  + voff + (size_t)srow * SEQ + kbase + scol;
        uint16_t* kl = &Ks[buf][srow][scol];
        uint16_t* vl = &Vs[buf][srow][scol];
        #pragma unroll
        for (int c = 0; c < 8; ++c) {
            g2l_b128(kg + c * 8, kl + c * 8);
            g2l_b128(vg + c * 8, vl + c * 8);
        }
    };

    // Q A-fragments over Dh (4 k-steps of 32), held in registers
    Frag qf[4];
    {
        const uint16_t* qrow = qkv + qoff + (size_t)(q0 + ln) * DHEAD;
        #pragma unroll
        for (int kk = 0; kk < 4; ++kk) {
            const uint16_t* p = qrow + kk * 32 + half * 8;
            qf[kk].u[0] = *(const uint4*)p;
            qf[kk].u[1] = *(const uint4*)(p + 16);
        }
    }

    f32x8 oAcc[8];
    #pragma unroll
    for (int t = 0; t < 8; ++t)
        #pragma unroll
        for (int e = 0; e < 8; ++e) oAcc[t][e] = 0.0f;
    float mst[8], lst[8];
    #pragma unroll
    for (int r = 0; r < 8; ++r) { mst[r] = -1e30f; lst[r] = 0.0f; }

    const float scale = 0.088388347648318447f;   // 1/sqrt(128)

    stageKV(0, 0);
    for (int kb = 0; kb <= qblk; ++kb) {
        const int p = kb & 1;
        wait_async0();
        __syncthreads();                          // K/V buffer p visible to all waves
        if (kb + 1 <= qblk) stageKV(kb + 1, p ^ 1);   // prefetch next block

        const int kbase = kb * 128;

        // ---- S = Q K^T : 8 key tiles x 4 k-steps, B-frags from LDS
        f32x8 sc[8];
        #pragma unroll
        for (int t = 0; t < 8; ++t)
            #pragma unroll
            for (int e = 0; e < 8; ++e) sc[t][e] = 0.0f;
        #pragma unroll
        for (int t = 0; t < 8; ++t) {
            #pragma unroll
            for (int kk = 0; kk < 4; ++kk) {
                Frag bfr;
                const uint16_t* pp = &Ks[p][t * 16 + ln][kk * 32 + half * 8];
                bfr.u[0] = *(const uint4*)pp;
                bfr.u[1] = *(const uint4*)(pp + 16);
                sc[t] = WMMA_BF16(qf[kk].v, bfr.v, sc[t]);
            }
        }

        // ---- causal mask on the diagonal block
        if (kb == qblk) {
            #pragma unroll
            for (int t = 0; t < 8; ++t) {
                int key = kbase + t * 16 + ln;
                #pragma unroll
                for (int r = 0; r < 8; ++r) {
                    int qr = q0 + r + 8 * half;
                    if (key > qr) sc[t][r] = -1e30f;
                }
            }
        }

        // ---- row max over 128 keys
        float pm[8];
        #pragma unroll
        for (int r = 0; r < 8; ++r) {
            float m = sc[0][r];
            #pragma unroll
            for (int t = 1; t < 8; ++t) m = fmaxf(m, sc[t][r]);
            m = fmaxf(m, swz_xor<1>(m));
            m = fmaxf(m, swz_xor<2>(m));
            m = fmaxf(m, swz_xor<4>(m));
            m = fmaxf(m, swz_xor<8>(m));
            pm[r] = m;
        }

        float al[8];
        #pragma unroll
        for (int r = 0; r < 8; ++r) {
            float mn = fmaxf(mst[r], pm[r] * scale);
            al[r] = __expf(mst[r] - mn);
            mst[r] = mn;
        }

        // ---- P = exp(S*scale - m); store bf16 to per-wave LDS; row sums
        float rs[8];
        #pragma unroll
        for (int r = 0; r < 8; ++r) rs[r] = 0.0f;
        #pragma unroll
        for (int t = 0; t < 8; ++t)
            #pragma unroll
            for (int r = 0; r < 8; ++r) {
                float pv = __expf(fmaf(sc[t][r], scale, -mst[r]));
                rs[r] += pv;
                Pl[wave][r + 8 * half][t * 16 + ln] = f2bf(pv);
            }
        #pragma unroll
        for (int r = 0; r < 8; ++r) {
            float s = rs[r];
            s += swz_xor<1>(s);
            s += swz_xor<2>(s);
            s += swz_xor<4>(s);
            s += swz_xor<8>(s);
            lst[r] = lst[r] * al[r] + s;
        }
        #pragma unroll
        for (int t = 0; t < 8; ++t)
            #pragma unroll
            for (int r = 0; r < 8; ++r) oAcc[t][r] *= al[r];

        // Pl is per-wave private; same-wave DS ops are in-order in HW.
        // Only prevent compile-time reordering of the P stores vs reloads.
        sched_mem_fence();

        // ---- O += P V : P A-frags from per-wave LDS, V B-frags from Vs[p]
        Frag pf[4];
        #pragma unroll
        for (int kk = 0; kk < 4; ++kk) {
            const uint16_t* pp = &Pl[wave][ln][kk * 32 + half * 8];
            pf[kk].u[0] = *(const uint4*)pp;
            pf[kk].u[1] = *(const uint4*)(pp + 16);
        }
        #pragma unroll
        for (int t = 0; t < 8; ++t) {
            #pragma unroll
            for (int kk = 0; kk < 4; ++kk) {
                Frag vf;
                const uint16_t* pp = &Vs[p][t * 16 + ln][kk * 32 + half * 8];
                vf.u[0] = *(const uint4*)pp;
                vf.u[1] = *(const uint4*)(pp + 16);
                oAcc[t] = WMMA_BF16(pf[kk].v, vf.v, oAcc[t]);
            }
        }
        // no bottom barrier: next iteration's top barrier orders reads of
        // buffer p against the stage() that overwrites it
    }

    // ---- finalize: O /= l, scatter bf16 to o_merged[b, s, h*128 + dh]
    float inv[8];
    #pragma unroll
    for (int r = 0; r < 8; ++r) inv[r] = 1.0f / lst[r];
    #pragma unroll
    for (int t = 0; t < 8; ++t)
        #pragma unroll
        for (int r = 0; r < 8; ++r) {
            int s   = q0 + r + 8 * half;
            int col = h * DHEAD + t * 16 + ln;
            om[((size_t)b * SEQ + s) * DMODEL + col] = f2bf(oAcc[t][r] * inv[r]);
        }
}

// ---------------------------------------------------------------------------
// Launcher
// ---------------------------------------------------------------------------
extern "C" void kernel_launch(void* const* d_in, const int* in_sizes, int n_in,
                              void* d_out, int out_size, void* d_ws, size_t ws_size,
                              hipStream_t stream) {
    const float* x       = (const float*)d_in[0];
    const float* w_qkv   = (const float*)d_in[1];
    const float* w_out   = (const float*)d_in[2];
    const int*   start_p = (const int*)d_in[3];
    float* out = (float*)d_out;

    char* ws = (char*)d_ws;
    uint16_t* XB  = (uint16_t*)ws; ws += (size_t)MROWS * DMODEL * 2;          // 16 MB
    uint16_t* WQ  = (uint16_t*)ws; ws += (size_t)3 * DMODEL * DMODEL * 2;     // 24 MB
    uint16_t* WO  = (uint16_t*)ws; ws += (size_t)DMODEL * DMODEL * 2;         // 8 MB
    uint16_t* QKV = (uint16_t*)ws; ws += (size_t)BATCH * 3 * DMODEL * SEQ * 2;// 48 MB
    uint16_t* VT  = (uint16_t*)ws; ws += (size_t)MROWS * DMODEL * 2;          // 16 MB
    uint16_t* OM  = (uint16_t*)ws;                                            // 16 MB

    const int nx = MROWS * DMODEL;          // 8388608
    const int nq = 3 * DMODEL * DMODEL;     // 12582912
    const int no = DMODEL * DMODEL;         // 4194304
    cvt_f32_bf16<<<(nx + 255) / 256, 256, 0, stream>>>(x, XB, nx);
    cvt_f32_bf16<<<(nq + 255) / 256, 256, 0, stream>>>(w_qkv, WQ, nq);
    cvt_f32_bf16<<<(no + 255) / 256, 256, 0, stream>>>(w_out, WO, no);

    // QKV projection: [4096,2048] x [6144,2048]^T -> scatter bf16 QKV
    gemm_bf16_nt<1><<<dim3(48, 32), 256, 0, stream>>>(XB, WQ, nullptr, QKV,
                                                      MROWS, 3 * DMODEL, DMODEL);

    rope_kernel<<<(1 << 23) / 256, 256, 0, stream>>>(QKV, start_p);
    vtrans_kernel<<<(1 << 23) / 256, 256, 0, stream>>>(QKV, VT);

    flash_attn_kernel<<<BATCH * NHEADS * (SEQ / 128), 256, 0, stream>>>(QKV, VT, OM);

    // Output projection: [4096,2048] x [2048,2048]^T -> fp32 d_out
    gemm_bf16_nt<0><<<dim3(16, 32), 256, 0, stream>>>(OM, WO, out, nullptr,
                                                      MROWS, DMODEL, DMODEL);
}
